// GNN_83356725281276
// MI455X (gfx1250) — matrix operand order repaired and checked by reading, not compiled, not run on hardware
//
#include <hip/hip_runtime.h>
#include <hip/hip_bf16.h>

#define NN 16384
#define CC 64
#define GG 128

typedef __attribute__((ext_vector_type(16))) __bf16 v16bf;
typedef __attribute__((ext_vector_type(8)))  __bf16 v8bf;
typedef __attribute__((ext_vector_type(8)))  float  v8f;

union BF16x16 { v16bf v; v8bf h[2]; unsigned u[8]; __bf16 e[16]; };
union F32x8   { v8f  v; float e[8]; };

// pack two fp32 into one dword of two bf16 (truncation) with a single v_perm_b32
__device__ __forceinline__ unsigned pack_bf16(float lo, float hi) {
  return __builtin_amdgcn_perm(__builtin_bit_cast(unsigned, hi),
                               __builtin_bit_cast(unsigned, lo),
                               0x07060302u);
}

// stage X[kb..kb+63][0..63] into LDS transposed (Bs[n][k]) as bf16.
// Each thread owns one column n and 16 consecutive k values -> 2 ds_store_b128.
__device__ __forceinline__ void stage_chunk(const float* __restrict__ X, int kb,
                                            __bf16 (*Bs)[72], int t) {
  const int n  = t & 63;
  const int k0 = (t >> 6) * 16;
  const float* src = X + (size_t)(kb + k0) * CC + n;
  float f[16];
  #pragma unroll
  for (int i = 0; i < 16; ++i) f[i] = src[(size_t)i * CC];
  unsigned u[8];
  #pragma unroll
  for (int i = 0; i < 8; ++i) u[i] = pack_bf16(f[2 * i], f[2 * i + 1]);
  uint4* dst = (uint4*)&Bs[n][k0];
  dst[0] = make_uint4(u[0], u[1], u[2], u[3]);
  dst[1] = make_uint4(u[4], u[5], u[6], u[7]);
}

// Out[N,64] = relu( ((adj + I) @ X) @ W^T + bias )
// identity folded as adj@X + X in the epilogue.
// One wave computes 16 output rows x 64 cols. 8 waves/WG -> 128 rows/WG.
__global__ __launch_bounds__(256) void prop_fc_kernel(
    const float* __restrict__ adj, const float* __restrict__ X,
    const float* __restrict__ W,   const float* __restrict__ bias,
    float* __restrict__ Out)
{
  __shared__ __bf16 Bs[2][64][72];  // double-buffered transposed X chunk
  __shared__ float  Ps[8][16][68];  // per-wave P tile staging for fused FC

  const int lane   = threadIdx.x & 31;
  const int wave   = threadIdx.x >> 5;
  const int half   = lane >> 4;        // 0: lanes 0-15, 1: lanes 16-31
  const int lanelo = lane & 15;
  const int waveRow = blockIdx.x * 128 + wave * 16;
  const int r = waveRow + lanelo;      // A-matrix row for this lane

  F32x8 acc[4];
  #pragma unroll
  for (int j = 0; j < 4; ++j)
    #pragma unroll
    for (int rr = 0; rr < 8; ++rr) acc[j].e[rr] = 0.0f;

  const float* ap = adj + (size_t)r * NN;

  stage_chunk(X, 0, Bs[0], threadIdx.x);
  __syncthreads();

  int buf = 0;
  for (int kb = 0; kb < NN; kb += 64) {
    if (kb + 64 < NN) stage_chunk(X, kb + 64, Bs[buf ^ 1], threadIdx.x);

    #pragma unroll
    for (int kk = 0; kk < 64; kk += 32) {
      // ---- A fragment: 16x32 bf16 slice of adj ----
      BF16x16 a;
      {
        const int k0 = kb + kk + half * 8;   // elements 0..7  -> K = k0..k0+7
        const int k1 = k0 + 16;              // elements 8..15 -> K = k1..k1+7
        float fa[8], fb[8];
        *(float4*)&fa[0] = *(const float4*)(ap + k0);
        *(float4*)&fa[4] = *(const float4*)(ap + k0 + 4);
        *(float4*)&fb[0] = *(const float4*)(ap + k1);
        *(float4*)&fb[4] = *(const float4*)(ap + k1 + 4);
        #pragma unroll
        for (int i = 0; i < 4; ++i) {
          a.u[i]     = pack_bf16(fa[2 * i], fa[2 * i + 1]);
          a.u[4 + i] = pack_bf16(fb[2 * i], fb[2 * i + 1]);
        }
      }
      // ---- 4 B fragments (col blocks) from LDS, 4 WMMAs ----
      #pragma unroll
      for (int j = 0; j < 4; ++j) {
        BF16x16 b;
        const __bf16* bp = &Bs[buf][j * 16 + lanelo][kk + half * 8];
        b.h[0] = *(const v8bf*)bp;
        b.h[1] = *(const v8bf*)(bp + 16);
        acc[j].v = __builtin_amdgcn_wmma_f32_16x16x32_bf16(
            false, a.v, false, b.v, (short)0, acc[j].v, false, false);
      }
    }
    __syncthreads();
    buf ^= 1;
  }

  // ---- fused FC epilogue: H = relu((P + X_tile) @ W^T + bias) ----
  // stage P tile (16x64 f32) into per-wave LDS region, folding the +I term
  #pragma unroll
  for (int j = 0; j < 4; ++j)
    #pragma unroll
    for (int rr = 0; rr < 8; ++rr) {
      const int m = rr + half * 8;
      const int n = j * 16 + lanelo;
      Ps[wave][m][n] = acc[j].e[rr] + X[(size_t)(waveRow + m) * CC + n];
    }

  F32x8 hacc[4];
  #pragma unroll
  for (int j = 0; j < 4; ++j) {
    const float bb = bias[j * 16 + lanelo];
    #pragma unroll
    for (int rr = 0; rr < 8; ++rr) hacc[j].e[rr] = bb;
  }

  #pragma unroll
  for (int kk = 0; kk < 64; kk += 32) {
    BF16x16 a;
    {
      const int base0 = kk + half * 16;
      const int base1 = base0 + 8;
      #pragma unroll
      for (int i = 0; i < 4; ++i) {
        a.u[i]     = pack_bf16(Ps[wave][lanelo][base0 + 2 * i],
                               Ps[wave][lanelo][base0 + 2 * i + 1]);
        a.u[4 + i] = pack_bf16(Ps[wave][lanelo][base1 + 2 * i],
                               Ps[wave][lanelo][base1 + 2 * i + 1]);
      }
    }
    #pragma unroll
    for (int j = 0; j < 4; ++j) {
      const int n = j * 16 + lanelo;            // output column
      const float* wp = W + (size_t)n * CC + kk + half * 16;
      float fw[16];
      *(float4*)&fw[0]  = *(const float4*)(wp + 0);
      *(float4*)&fw[4]  = *(const float4*)(wp + 4);
      *(float4*)&fw[8]  = *(const float4*)(wp + 8);
      *(float4*)&fw[12] = *(const float4*)(wp + 12);
      BF16x16 b;                                // B[k][n] = W[n][k], contiguous in k
      #pragma unroll
      for (int i = 0; i < 8; ++i) b.u[i] = pack_bf16(fw[2 * i], fw[2 * i + 1]);
      hacc[j].v = __builtin_amdgcn_wmma_f32_16x16x32_bf16(
          false, a.v, false, b.v, (short)0, hacc[j].v, false, false);
    }
  }

  #pragma unroll
  for (int j = 0; j < 4; ++j)
    #pragma unroll
    for (int rr = 0; rr < 8; ++rr) {
      float v = hacc[j].e[rr];
      v = v > 0.0f ? v : 0.0f;
      Out[(size_t)(waveRow + rr + half * 8) * CC + j * 16 + lanelo] = v;
    }
}

__global__ void zero_kernel(float* __restrict__ p, int n) {
  int i = blockIdx.x * blockDim.x + threadIdx.x;
  if (i < n) p[i] = 0.0f;
}

// segment_sum over sorted idx: each thread folds 16 rows of one column,
// flushing an atomicAdd only at graph-id boundaries.
__global__ void segsum_kernel(const float* __restrict__ X2,
                              const int* __restrict__ idx,
                              float* __restrict__ seg) {
  const int t   = blockIdx.x * blockDim.x + threadIdx.x;  // 0..65535
  const int col = t & 63;
  const int r0  = (t >> 6) * 16;
  float local = 0.0f;
  int cur = idx[r0];
  for (int r = r0; r < r0 + 16; ++r) {
    int id = idx[r];
    if (id != cur) {
      atomicAdd(&seg[cur * CC + col], local);
      local = 0.0f; cur = id;
    }
    local += X2[(size_t)r * CC + col];
  }
  atomicAdd(&seg[cur * CC + col], local);
}

// one thread per graph: relu(seg@W3^T+b3) -> @W4^T+b4 -> log_softmax
__global__ __launch_bounds__(128) void head_kernel(
    const float* __restrict__ seg,
    const float* __restrict__ W3, const float* __restrict__ b3,
    const float* __restrict__ W4, const float* __restrict__ b4,
    float* __restrict__ out) {
  const int g = threadIdx.x;
  if (g >= GG) return;
  float s[64];
  #pragma unroll
  for (int i = 0; i < 64; ++i) s[i] = seg[g * CC + i];
  float o[64];
  for (int h = 0; h < 64; ++h) {
    float a = b3[h];
    const float* w = W3 + h * 64;
    #pragma unroll
    for (int k = 0; k < 64; ++k) a += s[k] * w[k];
    o[h] = a > 0.0f ? a : 0.0f;
  }
  float l[10];
  #pragma unroll
  for (int c = 0; c < 10; ++c) {
    float a = b4[c];
    const float* w = W4 + c * 64;
    #pragma unroll
    for (int k = 0; k < 64; ++k) a += o[k] * w[k];
    l[c] = a;
  }
  float m = l[0];
  #pragma unroll
  for (int c = 1; c < 10; ++c) m = l[c] > m ? l[c] : m;
  float sum = 0.0f;
  #pragma unroll
  for (int c = 0; c < 10; ++c) sum += __expf(l[c] - m);
  const float ls = __logf(sum);
  #pragma unroll
  for (int c = 0; c < 10; ++c) out[g * 10 + c] = l[c] - m - ls;
}

extern "C" void kernel_launch(void* const* d_in, const int* in_sizes, int n_in,
                              void* d_out, int out_size, void* d_ws, size_t ws_size,
                              hipStream_t stream) {
  const float* x_in = (const float*)d_in[0];
  const float* adj  = (const float*)d_in[1];
  const int*   idx  = (const int*)d_in[2];
  const float* W1 = (const float*)d_in[3];
  const float* b1 = (const float*)d_in[4];
  const float* W2 = (const float*)d_in[5];
  const float* b2 = (const float*)d_in[6];
  const float* W3 = (const float*)d_in[7];
  const float* b3 = (const float*)d_in[8];
  const float* W4 = (const float*)d_in[9];
  const float* b4 = (const float*)d_in[10];
  float* out = (float*)d_out;

  float* H   = (float*)d_ws;                 // [N,64]
  float* X2  = H  + (size_t)NN * CC;         // [N,64]
  float* seg = X2 + (size_t)NN * CC;         // [128,64]

  // layer 1+2: P1 = (adj+I)@x_in ; H = relu(P1@W1^T + b1)
  prop_fc_kernel<<<NN / 128, 256, 0, stream>>>(adj, x_in, W1, b1, H);
  // layer 3+4: P2 = (adj+I)@H ; X2 = relu(P2@W2^T + b2)
  prop_fc_kernel<<<NN / 128, 256, 0, stream>>>(adj, H, W2, b2, X2);
  // readout
  zero_kernel<<<(GG * CC + 255) / 256, 256, 0, stream>>>(seg, GG * CC);
  segsum_kernel<<<(NN / 16 * CC) / 256, 256, 0, stream>>>(X2, idx, seg);
  head_kernel<<<1, 128, 0, stream>>>(seg, W3, b3, W4, b4, out);
}
